// EdgeConvLayer_6992206757905
// MI455X (gfx1250) — matrix operand order repaired and checked by reading, not compiled, not run on hardware
//
#include <hip/hip_runtime.h>

// ---------------------------------------------------------------------------
// EdgeConv (fully-connected) for MI455X / gfx1250, wave32 + WMMA f16->f32.
//   B=32, N=128, F=64, widths 128/128/128, leaky_relu(0.2)
// Layer 1 is decomposed (concat structure): U = f*W1_top + b1, V = f*W1_bot.
// Layers 2/3 are per-edge GEMMs done with v_wmma_f32_16x16x32_f16.
// ---------------------------------------------------------------------------

typedef _Float16 v16h __attribute__((ext_vector_type(16)));
typedef _Float16 v8h  __attribute__((ext_vector_type(8)));
typedef float    v8f  __attribute__((ext_vector_type(8)));

#define NB   32
#define NN   128
#define NF   64
#define CH   128
#define SROW 136   // padded LDS row stride in halves (272B = 16B multiple)

__device__ __forceinline__ float lrelu(float x) { return x >= 0.f ? x : 0.2f * x; }

__device__ __forceinline__ v16h cat16(v8h lo, v8h hi) {
    return __builtin_shufflevector(lo, hi, 0, 1, 2, 3, 4, 5, 6, 7,
                                           8, 9, 10, 11, 12, 13, 14, 15);
}

// ---------------------------------------------------------------------------
// Prep 1: W2/W3 -> transposed f16 ([out][in], contiguous over K for B-frags),
// plus per-batch mask sums.
// ---------------------------------------------------------------------------
__global__ void prep_weights(const float* __restrict__ W2,
                             const float* __restrict__ W3,
                             const unsigned char* __restrict__ mask,
                             _Float16* __restrict__ W2T,
                             _Float16* __restrict__ W3T,
                             float* __restrict__ msum) {
    int idx = blockIdx.x * 256 + threadIdx.x;   // grid 64 -> 16384
    int o = idx >> 7, k = idx & 127;
    W2T[idx] = (_Float16)W2[k * CH + o];
    W3T[idx] = (_Float16)W3[k * CH + o];
    if (blockIdx.x == 0 && threadIdx.x < NB) {
        float s = 0.f;
        for (int j = 0; j < NN; ++j) s += (float)mask[threadIdx.x * NN + j];
        msum[threadIdx.x] = s;
    }
}

// ---------------------------------------------------------------------------
// Prep 2: U[row,c] = sum_f feat[row,f]*W1[f,c] + b1[c]
//         V[row,c] = sum_f feat[row,f]*W1[64+f,c]      (row = b*N + node)
// fp32 throughout; tiny (0.13 GFLOP total).
// ---------------------------------------------------------------------------
__global__ void prep_uv(const float* __restrict__ feat,
                        const float* __restrict__ W1,
                        const float* __restrict__ b1,
                        float* __restrict__ U, float* __restrict__ V) {
    int row = blockIdx.x;     // 0..4095
    int c   = threadIdx.x;    // 0..127
    __shared__ float f[NF];
    if (c < NF) f[c] = feat[row * NF + c];
    __syncthreads();
    float u = b1[c], v = 0.f;
#pragma unroll 8
    for (int k = 0; k < NF; ++k) {
        float x = f[k];
        u = fmaf(x, W1[k * CH + c], u);
        v = fmaf(x, W1[(NF + k) * CH + c], v);
    }
    U[row * CH + c] = u;
    V[row * CH + c] = v;
}

// ---------------------------------------------------------------------------
// One WMMA pass: 16-row A-block (this wave's rows in LDS) x WT[128x128 f16]
// A-frag (16-bit A 16x32): lane L -> row m0+(L&15), 16 contiguous K at
// koff, koff = kb + (L<16 ? 0 : 16).
// B-frag (16-bit B 32x16): lane L -> col 16n+(L&15), 16 contiguous K at
// kb + (L<16 ? 0 : 16)  (WT stored [out][in] so this is one contiguous run).
// ---------------------------------------------------------------------------
__device__ __forceinline__ void gemm_rowblock(const _Float16* __restrict__ buf,
                                              const _Float16* __restrict__ WT,
                                              int arow, int akoff, int bkoff,
                                              int ncol, v8f acc[8]) {
#pragma unroll
    for (int ks = 0; ks < 4; ++ks) {
        int kb = ks * 32;
        v8h alo = *(const v8h*)&buf[arow * SROW + kb + akoff];
        v8h ahi = *(const v8h*)&buf[arow * SROW + kb + akoff + 8];
        v16h a = cat16(alo, ahi);
#pragma unroll
        for (int n = 0; n < 8; ++n) {
            const _Float16* wp = WT + (16 * n + ncol) * CH + kb + bkoff;
            v8h blo = *(const v8h*)wp;
            v8h bhi = *(const v8h*)(wp + 8);
            v16h bb = cat16(blo, bhi);
            acc[n] = __builtin_amdgcn_wmma_f32_16x16x32_f16(
                false, a, false, bb, (short)0, acc[n], false, false);
        }
    }
}

// ---------------------------------------------------------------------------
// Main kernel: one workgroup per (b,i). 8 waves; wave w owns j-rows
// [16w,16w+16). h1 built in LDS (f16), layers 2/3 in-place per wave,
// masked mean via ds_add_f32 reduction, multiply_no_nan + final lrelu.
// ---------------------------------------------------------------------------
__global__ __launch_bounds__(256) void edgeconv_main(
    const float* __restrict__ U, const float* __restrict__ V,
    const _Float16* __restrict__ W2T, const _Float16* __restrict__ W3T,
    const float* __restrict__ b2, const float* __restrict__ b3,
    const unsigned char* __restrict__ mask, const float* __restrict__ msum,
    float* __restrict__ out) {
    __shared__ _Float16 buf[NN * SROW];   // 34816 B: h1 -> h2 (in place per wave)
    __shared__ float    ssum[CH];         // channel accumulators

    const int b = blockIdx.y, i = blockIdx.x;
    const int t = threadIdx.x;
    const float* Urow = U + (b * NN + i) * CH;
    const float* Vb   = V + b * NN * CH;

    // h1[j,c] = lrelu(U[b,i,c] + V[b,j,c])  (b1 folded into U)
    for (int idx = t; idx < NN * CH; idx += 256) {
        int j = idx >> 7, c = idx & 127;
        buf[j * SROW + c] = (_Float16)lrelu(Urow[c] + Vb[idx]);
    }
    if (t < CH) ssum[t] = 0.f;
    __syncthreads();

    const int wave  = t >> 5, lane = t & 31;
    const int m0    = wave * 16;
    const int arow  = m0 + (lane & 15);
    const int akoff = (lane < 16) ? 0 : 16;
    const int bkoff = (lane < 16) ? 0 : 16;
    const int ncol  = lane & 15;
    const int rbase = m0 + ((lane < 16) ? 0 : 8);   // C/D tile: M = r / r+8

    // ---- layer 2: h1 x W2 + b2, lrelu, store back to this wave's rows ----
    v8f acc[8];
#pragma unroll
    for (int n = 0; n < 8; ++n) acc[n] = (v8f){};
    gemm_rowblock(buf, W2T, arow, akoff, bkoff, ncol, acc);
#pragma unroll
    for (int n = 0; n < 8; ++n) {
        int col = 16 * n + ncol;
        float bias = b2[col];
#pragma unroll
        for (int r = 0; r < 8; ++r)
            buf[(rbase + r) * SROW + col] = (_Float16)lrelu(acc[n][r] + bias);
    }
    __syncthreads();   // wave-local in-place is safe; barrier for hygiene

    // ---- layer 3: h2 x W3 + b3, lrelu, masked row-reduce over j ----
#pragma unroll
    for (int n = 0; n < 8; ++n) acc[n] = (v8f){};
    gemm_rowblock(buf, W3T, arow, akoff, bkoff, ncol, acc);

    float wrow[8];
#pragma unroll
    for (int r = 0; r < 8; ++r)
        wrow[r] = (float)mask[b * NN + rbase + r];   // m[b,j]
#pragma unroll
    for (int n = 0; n < 8; ++n) {
        int col = 16 * n + ncol;
        float bias = b3[col];
        float p = 0.f;
#pragma unroll
        for (int r = 0; r < 8; ++r)
            p += wrow[r] * lrelu(acc[n][r] + bias);
        atomicAdd(&ssum[col], p);                    // ds_add_f32
    }
    __syncthreads();

    if (t < CH) {
        float mi  = (float)mask[b * NN + i];
        float num = mi * ssum[t];                    // sum_j m_i m_j h3[j,t]
        float den = mi * msum[b];                    // sum_j m_i m_j
        float o = (num == 0.f) ? 0.f : num / ((den == 0.f) ? 1.f : den);
        out[(b * NN + i) * CH + t] = lrelu(o);
    }
}

// ---------------------------------------------------------------------------
// Host-side launcher. Inputs: features, mask, W1, b1, W2, b2, W3, b3.
// Workspace: U(2MB) | V(2MB) | W2T(32KB) | W3T(32KB) | msum(128B)
// ---------------------------------------------------------------------------
extern "C" void kernel_launch(void* const* d_in, const int* in_sizes, int n_in,
                              void* d_out, int out_size, void* d_ws, size_t ws_size,
                              hipStream_t stream) {
    const float*         feat = (const float*)d_in[0];
    const unsigned char* mask = (const unsigned char*)d_in[1];  // jax bool -> 1B
    const float*         W1   = (const float*)d_in[2];
    const float*         b1   = (const float*)d_in[3];
    const float*         W2   = (const float*)d_in[4];
    const float*         b2   = (const float*)d_in[5];
    const float*         W3   = (const float*)d_in[6];
    const float*         b3   = (const float*)d_in[7];
    float*               out  = (float*)d_out;

    char* ws = (char*)d_ws;
    float*     U    = (float*)(ws + 0);
    float*     V    = (float*)(ws + 2097152);
    _Float16*  W2T  = (_Float16*)(ws + 4194304);
    _Float16*  W3T  = (_Float16*)(ws + 4227072);
    float*     msum = (float*)(ws + 4259840);

    prep_weights<<<64, 256, 0, stream>>>(W2, W3, mask, W2T, W3T, msum);
    prep_uv<<<NB * NN, CH, 0, stream>>>(feat, W1, b1, U, V);
    edgeconv_main<<<dim3(NN, NB), 256, 0, stream>>>(U, V, W2T, W3T, b2, b3,
                                                    mask, msum, out);
}
